// GaussianEmbedding_22874995818498
// MI455X (gfx1250) — matrix operand order repaired and checked by reading, not compiled, not run on hardware
//
#include <hip/hip_runtime.h>
#include <hip/hip_bf16.h>

#define BB   32
#define TC   256   // real char positions
#define TC2  258   // chars incl. 2 trailing pad columns
#define TCP  288   // K padded to multiple of 32 (9 * 32)
#define DEMB 384
#define NKB  (TCP / 32)   // 9 K-chunks

typedef __attribute__((ext_vector_type(16))) _Float16 v16h;
typedef __attribute__((ext_vector_type(8)))  _Float16 v8h;
typedef __attribute__((ext_vector_type(8)))  float    v8f;

static __device__ __forceinline__ v8f wmma_f16(v16h a, v16h b, v8f c) {
  return __builtin_amdgcn_wmma_f32_16x16x32_f16(false, a, false, b, (short)0, c,
                                                false, false);
}

// ---------------- per-batch prep: centers, 1/sigma, pdf coefficient ----------
__global__ void ga_prep(const int* __restrict__ durs, float* __restrict__ center,
                        float* __restrict__ invsig, float* __restrict__ coef,
                        int* __restrict__ tdur) {
  int b = threadIdx.x;
  if (b >= BB) return;
  float csum = 0.f;
  for (int c = 0; c < TCP; ++c) {
    float d = (c < TC) ? (float)durs[b * TC + c] : 0.f;
    float ctr = 0.5f * d + csum;
    csum += d;
    float sig = d * 0.5f + 1e-6f;            // d / SIGMA_C + EPS, SIGMA_C = 2
    float inv = 1.0f / sig;
    float cf  = (c < TC) ? inv * 0.3989422804014327f : 0.f;  // exp(-log sig - log sqrt(2pi))
    if (c >= TC2) { ctr = 0.f; inv = 0.f; }  // padded K columns -> w = cf*exp(0) = 0
    center[b * TCP + c] = ctr;
    invsig[b * TCP + c] = inv;
    coef  [b * TCP + c] = cf;
  }
  tdur[b] = (int)(csum + 0.5f);
}

// ---- gather + transpose embeddings to f16: EgT[b][d][c], c contiguous -------
__global__ void ga_gather(const int* __restrict__ text, const float* __restrict__ embed,
                          _Float16* __restrict__ egt) {
  __shared__ int tok[TCP];
  int b = blockIdx.x;
  for (int c = threadIdx.x; c < TCP; c += blockDim.x)
    tok[c] = (c < TC) ? text[b * TC + c] : 0;   // trailing pads use token 0 (PAD row)
  __syncthreads();
  for (int idx = threadIdx.x; idx < DEMB * TCP; idx += blockDim.x) {
    int d = idx / TCP, c = idx - d * TCP;
    float v = (c < TC2) ? embed[tok[c] * DEMB + d] : 0.f;
    egt[(size_t)b * DEMB * TCP + idx] = (_Float16)v;
  }
}

// ---------------- main: weights in LDS, then WMMA f16 GEMM -------------------
__global__ __launch_bounds__(256) void ga_main(
    const float* __restrict__ center, const float* __restrict__ invsig,
    const float* __restrict__ coef, const int* __restrict__ tdur,
    const _Float16* __restrict__ egt, float* __restrict__ out, int Tt) {
  __shared__ float wf[16 * TCP];                         // f32 weights
  __shared__ __align__(32) _Float16 wh[16 * TCP];        // normalized f16 weights
  __shared__ float psum[256];
  __shared__ float rnorm[16];
  __shared__ int   padf[16];

  const int b   = blockIdx.y;
  const int t0  = blockIdx.x * 16;
  const int tid = threadIdx.x;
  const int td  = tdur[b];

  const float* ctr = center + b * TCP;
  const float* inv = invsig + b * TCP;
  const float* cf  = coef   + b * TCP;

  // Gaussian pdf tile [16][TCP]
  for (int idx = tid; idx < 16 * TCP; idx += 256) {
    int tl = idx / TCP, c = idx - tl * TCP;
    float t = (float)(t0 + tl) + 0.5f;
    float z = (t - ctr[c]) * inv[c];
    wf[idx] = cf[c] * __expf(-0.5f * z * z);
  }
  __syncthreads();

  // row sums (16 rows x 16 partials of 18 cols)
  {
    int row = tid >> 4, part = tid & 15;
    float s = 0.f;
    int base = row * TCP + part * 18;
    for (int j = 0; j < 18; ++j) s += wf[base + j];
    psum[row * 16 + part] = s;
  }
  __syncthreads();
  if (tid < 16) {
    float s = 0.f;
    for (int j = 0; j < 16; ++j) s += psum[tid * 16 + j];
    rnorm[tid] = 1.0f / (s + 1e-6f);
    padf[tid]  = ((t0 + tid) >= td) ? 1 : 0;
  }
  __syncthreads();

  // normalize + time-pad handling, convert to f16
  for (int idx = tid; idx < 16 * TCP; idx += 256) {
    int tl = idx / TCP, c = idx - tl * TCP;
    float v = padf[tl] ? ((c == TC2 - 1) ? 1.0f : 0.0f) : wf[idx] * rnorm[tl];
    wh[idx] = (_Float16)v;
  }
  __syncthreads();

  const int lane = tid & 31, wid = tid >> 5;
  const int m = lane & 15, g = lane >> 4;

  // Preload all 9 A fragments (contiguous K order: elements 0-15 = K g*16..+15
  // for row m, matching B) — reused by all 3 N-tiles.
  v16h afrag[NKB];
#pragma unroll
  for (int kb = 0; kb < NKB; ++kb) {
    const _Float16* ap = &wh[m * TCP + kb * 32 + g * 16];
    v8h alo = *(const v8h*)ap;
    v8h ahi = *(const v8h*)(ap + 8);
    afrag[kb] = __builtin_shufflevector(alo, ahi,
                  0,1,2,3,4,5,6,7,8,9,10,11,12,13,14,15);
  }

  // Each wave owns N-tiles wid, wid+8, wid+16 — processed concurrently
  // (3 independent WMMA chains, 6 b128 loads in flight per K-step).
  const int col0 = wid * 16 + m;                 // D columns: col0, +128, +256
  const _Float16* bp0 = egt + ((size_t)b * DEMB + col0) * TCP + g * 16;
  const _Float16* bp1 = bp0 + (size_t)128 * TCP;
  const _Float16* bp2 = bp1 + (size_t)128 * TCP;

  v8f acc0 = {}, acc1 = {}, acc2 = {};
  v16h f0 = *(const v16h*)bp0;
  v16h f1 = *(const v16h*)bp1;
  v16h f2 = *(const v16h*)bp2;
#pragma unroll
  for (int kb = 0; kb < NKB; ++kb) {
    v16h n0 = f0, n1 = f1, n2 = f2;
    if (kb < NKB - 1) {                          // prefetch next K-chunk
      n0 = *(const v16h*)(bp0 + (kb + 1) * 32);
      n1 = *(const v16h*)(bp1 + (kb + 1) * 32);
      n2 = *(const v16h*)(bp2 + (kb + 1) * 32);
    }
    acc0 = wmma_f16(afrag[kb], f0, acc0);
    acc1 = wmma_f16(afrag[kb], f1, acc1);
    acc2 = wmma_f16(afrag[kb], f2, acc2);
    f0 = n0; f1 = n1; f2 = n2;
  }

  // C/D layout: VGPR r -> row (r + 8*g), lane (l&15) -> column
  const int trow0 = t0 + 8 * g;
  float* orow = out + ((size_t)b * Tt + trow0) * DEMB;
  if (t0 + 16 <= Tt) {                           // block-uniform fast path
#pragma unroll
    for (int r = 0; r < 8; ++r) {
      float* p = orow + (size_t)r * DEMB;
      p[col0] = acc0[r];
      p[col0 + 128] = acc1[r];
      p[col0 + 256] = acc2[r];
    }
  } else {
#pragma unroll
    for (int r = 0; r < 8; ++r) {
      if (trow0 + r < Tt) {
        float* p = orow + (size_t)r * DEMB;
        p[col0] = acc0[r];
        p[col0 + 128] = acc1[r];
        p[col0 + 256] = acc2[r];
      }
    }
  }
}

extern "C" void kernel_launch(void* const* d_in, const int* in_sizes, int n_in,
                              void* d_out, int out_size, void* d_ws, size_t ws_size,
                              hipStream_t stream) {
  (void)in_sizes; (void)n_in; (void)ws_size;
  const int*   text  = (const int*)d_in[0];
  const int*   durs  = (const int*)d_in[1];
  const float* embed = (const float*)d_in[2];
  // d_in[3] is total_time on device; host-side grid sizing derives it from out_size
  int Tt = out_size / (BB * DEMB);

  char* ws = (char*)d_ws;
  float*    center = (float*)(ws + 0);
  float*    invsig = (float*)(ws + 36864);
  float*    coef   = (float*)(ws + 73728);
  int*      tdur   = (int*)  (ws + 110592);
  _Float16* egt    = (_Float16*)(ws + 110720);   // 32*384*288 f16 ~= 6.8 MB

  ga_prep<<<1, 32, 0, stream>>>(durs, center, invsig, coef, tdur);
  ga_gather<<<BB, 256, 0, stream>>>(text, embed, egt);
  dim3 grid((Tt + 15) / 16, BB);
  ga_main<<<grid, 256, 0, stream>>>(center, invsig, coef, tdur, egt, (float*)d_out, Tt);
}
